// TransformerGraphEncoder_83923660963961
// MI455X (gfx1250) — matrix operand order repaired and checked by reading, not compiled, not run on hardware
//
#include <hip/hip_runtime.h>
#include <math.h>

// ---------------------------------------------------------------------------
// TransformerGraphEncoder forward for MI455X (gfx1250, wave32).
// GEMMs (~95% of 190 GFLOP) run on v_wmma_f32_16x16x32_bf16 with fp32
// accumulation. Wave tile 32x32: 4 WMMA per 32-wide k-step at 2.0 b128
// loads/WMMA. The k-loop is manually unrolled x2 with ping-pong fragment
// register sets (96 VGPRs total for acc+frags) so prefetch loads overlap the
// WMMAs of the other set -- no register rotation copies, no spills.
// Weights (~2.4MB bf16) stay resident in L2. N=22 attention on VALU + LDS.
// ---------------------------------------------------------------------------

typedef __attribute__((ext_vector_type(16))) __bf16 v16bf;
typedef __attribute__((ext_vector_type(8)))  float  v8f;

#define B_   8
#define L_   256
#define N_   22
#define C_   128
#define H_   8
#define DK_  16
#define DFF_ 512
#define NL_  6
#define T_   (B_ * L_ * N_)   // 45056 tokens (divisible by 256)

__device__ __forceinline__ float wave_sum(float v) {
#pragma unroll
  for (int off = 16; off > 0; off >>= 1) v += __shfl_xor(v, off, 32);
  return v;
}

// A fragment (16x32 bf16): elems 0..7 at p, elems 8..15 at p+16 (per-lane).
__device__ __forceinline__ v16bf load_fragA(const __bf16* p) {
  union { struct { int4 lo, hi; } s; v16bf v; } u;
  u.s.lo = *reinterpret_cast<const int4*>(p);
  u.s.hi = *reinterpret_cast<const int4*>(p + 16);
  return u.v;
}
// B fragment (32x16 bf16): elems 0..7 at p, elems 8..15 at p+16 (per-lane).
__device__ __forceinline__ v16bf load_fragB(const __bf16* p) {
  union { struct { int4 lo, hi; } s; v16bf v; } u;
  u.s.lo = *reinterpret_cast<const int4*>(p);
  u.s.hi = *reinterpret_cast<const int4*>(p + 16);
  return u.v;
}

#define WMMA_BF16(a, b, c) \
  __builtin_amdgcn_wmma_f32_16x16x32_bf16(false, (a), false, (b), (short)0, (c), false, false)

// ---------------------------------------------------------------------------
// x_out = x + LN(x + pe, pe_g, pe_b)      (one wave per token, 4 ch / lane)
// ---------------------------------------------------------------------------
__global__ void pe_ln_kernel(const float* __restrict__ x,
                             const float* __restrict__ g,
                             const float* __restrict__ b,
                             float* __restrict__ out) {
  int wave = threadIdx.x >> 5, lane = threadIdx.x & 31;
  int t = blockIdx.x * 8 + wave;
  if (t >= T_) return;
  int l = (t / N_) % L_;

  float4 xv = reinterpret_cast<const float4*>(x + (size_t)t * C_)[lane];
  float xa[4] = {xv.x, xv.y, xv.z, xv.w};
  float z[4];
#pragma unroll
  for (int q = 0; q < 4; q++) {
    int c = lane * 4 + q;
    float freq = __powf(10000.0f, 2.0f * (float)c / (float)C_);
    float ang  = (float)l / freq;
    float pe   = (c & 1) ? __cosf(ang) : __sinf(ang);
    z[q] = xa[q] + pe;
  }
  float mean = wave_sum(z[0] + z[1] + z[2] + z[3]) * (1.0f / C_);
  float vs = 0.0f;
#pragma unroll
  for (int q = 0; q < 4; q++) { float d = z[q] - mean; vs += d * d; }
  float inv = 1.0f / sqrtf(wave_sum(vs) * (1.0f / C_) + 1e-5f);

  float4 ov;
  float* op = (float*)&ov;
#pragma unroll
  for (int q = 0; q < 4; q++) {
    int c = lane * 4 + q;
    op[q] = xa[q] + (z[q] - mean) * inv * g[c] + b[c];
  }
  reinterpret_cast<float4*>(out + (size_t)t * C_)[lane] = ov;
}

// ---------------------------------------------------------------------------
// out = LN(in (+ res), g, b); writes f32 and/or bf16 copies (bf16 feeds WMMA)
// In-place safe: all reads complete (into regs) before writes.
// ---------------------------------------------------------------------------
__global__ void ln_kernel(const float* __restrict__ in,
                          const float* __restrict__ res,   // nullable
                          const float* __restrict__ g,
                          const float* __restrict__ b,
                          float*  __restrict__ out32,      // nullable
                          __bf16* __restrict__ outbf) {    // nullable
  int wave = threadIdx.x >> 5, lane = threadIdx.x & 31;
  int t = blockIdx.x * 8 + wave;
  if (t >= T_) return;

  float4 xv = reinterpret_cast<const float4*>(in + (size_t)t * C_)[lane];
  float z[4] = {xv.x, xv.y, xv.z, xv.w};
  if (res) {
    float4 rv = reinterpret_cast<const float4*>(res + (size_t)t * C_)[lane];
    z[0] += rv.x; z[1] += rv.y; z[2] += rv.z; z[3] += rv.w;
  }
  float mean = wave_sum(z[0] + z[1] + z[2] + z[3]) * (1.0f / C_);
  float vs = 0.0f;
#pragma unroll
  for (int q = 0; q < 4; q++) { float d = z[q] - mean; vs += d * d; }
  float inv = 1.0f / sqrtf(wave_sum(vs) * (1.0f / C_) + 1e-5f);

  float v[4];
#pragma unroll
  for (int q = 0; q < 4; q++) {
    int c = lane * 4 + q;
    v[q] = (z[q] - mean) * inv * g[c] + b[c];
  }
  if (out32) {
    float4 ov; float* op = (float*)&ov;
#pragma unroll
    for (int q = 0; q < 4; q++) op[q] = v[q];
    reinterpret_cast<float4*>(out32 + (size_t)t * C_)[lane] = ov;
  }
  if (outbf) {
    union { __bf16 h[4]; uint2 u; } pk;
#pragma unroll
    for (int q = 0; q < 4; q++) pk.h[q] = (__bf16)v[q];
    reinterpret_cast<uint2*>(outbf + (size_t)t * C_)[lane] = pk.u;
  }
}

// ---------------------------------------------------------------------------
// f32 -> bf16 (weights, once per call; ~1.2M elements total)
// ---------------------------------------------------------------------------
__global__ void f2bf_kernel(const float* __restrict__ src,
                            __bf16* __restrict__ dst, int n) {
  int i = blockIdx.x * blockDim.x + threadIdx.x;
  if (i < n) dst[i] = (__bf16)src[i];
}

// ---------------------------------------------------------------------------
// D[M,N] = A[M,K](bf16) * W[N,K]^T(bf16) + bias, via v_wmma_f32_16x16x32_bf16.
// Wave: 32x32 tile. Block: 8 waves stacked in M = 256x32.
// k-loop unrolled x2 with ping-pong fragment sets X/Y (K % 64 == 0 always):
//   load Y(k+32); WMMA(X=k); load X(k+64); WMMA(Y=k+32); ...
// MODE 0: write f32.  MODE 1: bias + mish -> write bf16 (FF1 epilogue).
// Fragment layouts per CDNA5 ISA 7.12.2 (wave32):
//   A 16x32 bf16: lane<16 -> M=lane, elems 0..7 = K k..k+7, 8..15 = K k+16..23
//   B 32x16 bf16: lane<16 -> N=lane, elems 0..7 = K k..k+7, 8..15 = K k+16..23 ; lane>=16 -> k+8..15, k+24..31
//   C/D f32: elem e -> M = mtile + (lane>>4)*8 + e, N = ntile + (lane&15).
// ---------------------------------------------------------------------------
template <int MODE>
__global__ void gemm_bf16_kernel(const __bf16* __restrict__ A,
                                 const __bf16* __restrict__ W,
                                 const float* __restrict__ bias,
                                 float*  __restrict__ Df,
                                 __bf16* __restrict__ Dbf,
                                 int M, int Nn, int K) {
  int wave = threadIdx.x >> 5, lane = threadIdx.x & 31;
  int hf = lane >> 4, r = lane & 15;
  int mbase = blockIdx.y * 256 + wave * 32;
  int nbase = blockIdx.x * 32;
  if (mbase >= M) return;

  v8f acc00 = (v8f){0.f,0.f,0.f,0.f,0.f,0.f,0.f,0.f};
  v8f acc01 = acc00, acc10 = acc00, acc11 = acc00;

  const __bf16* a0 = A + (size_t)(mbase + r) * K + hf * 8;
  const __bf16* a1 = a0 + (size_t)16 * K;
  const __bf16* b0 = W + (size_t)(nbase + r) * K + hf * 8;
  const __bf16* b1 = b0 + (size_t)16 * K;

  // set X = k-step 0
  v16bf xa0 = load_fragA(a0), xa1 = load_fragA(a1);
  v16bf xb0 = load_fragB(b0), xb1 = load_fragB(b1);
  v16bf ya0, ya1, yb0, yb1;

  int k = 0;
  for (; k + 64 < K; k += 64) {
    // prefetch set Y = k+32 while computing on X
    ya0 = load_fragA(a0 + k + 32); ya1 = load_fragA(a1 + k + 32);
    yb0 = load_fragB(b0 + k + 32); yb1 = load_fragB(b1 + k + 32);
    acc00 = WMMA_BF16(xa0, xb0, acc00);
    acc01 = WMMA_BF16(xa0, xb1, acc01);
    acc10 = WMMA_BF16(xa1, xb0, acc10);
    acc11 = WMMA_BF16(xa1, xb1, acc11);
    // prefetch set X = k+64 while computing on Y
    xa0 = load_fragA(a0 + k + 64); xa1 = load_fragA(a1 + k + 64);
    xb0 = load_fragB(b0 + k + 64); xb1 = load_fragB(b1 + k + 64);
    acc00 = WMMA_BF16(ya0, yb0, acc00);
    acc01 = WMMA_BF16(ya0, yb1, acc01);
    acc10 = WMMA_BF16(ya1, yb0, acc10);
    acc11 = WMMA_BF16(ya1, yb1, acc11);
  }
  // tail: X holds k, Y gets the final k+32 step
  ya0 = load_fragA(a0 + k + 32); ya1 = load_fragA(a1 + k + 32);
  yb0 = load_fragB(b0 + k + 32); yb1 = load_fragB(b1 + k + 32);
  acc00 = WMMA_BF16(xa0, xb0, acc00);
  acc01 = WMMA_BF16(xa0, xb1, acc01);
  acc10 = WMMA_BF16(xa1, xb0, acc10);
  acc11 = WMMA_BF16(xa1, xb1, acc11);
  acc00 = WMMA_BF16(ya0, yb0, acc00);
  acc01 = WMMA_BF16(ya0, yb1, acc01);
  acc10 = WMMA_BF16(ya1, yb0, acc10);
  acc11 = WMMA_BF16(ya1, yb1, acc11);

  // epilogue
  v8f accs[2][2] = {{acc00, acc01}, {acc10, acc11}};
#pragma unroll
  for (int nt = 0; nt < 2; nt++) {
    int n = nbase + nt * 16 + r;
    float bv = bias[n];
#pragma unroll
    for (int mt = 0; mt < 2; mt++) {
#pragma unroll
      for (int e = 0; e < 8; e++) {
        int m = mbase + mt * 16 + hf * 8 + e;
        float val = accs[mt][nt][e] + bv;
        if (MODE == 1) {
          float sp = (val > 20.0f) ? val : log1pf(__expf(val));
          Dbf[(size_t)m * Nn + n] = (__bf16)(val * tanhf(sp));
        } else {
          Df[(size_t)m * Nn + n] = val;
        }
      }
    }
  }
}

// ---------------------------------------------------------------------------
// Attention over nodes (N=22) per (b,l,h). One wave per head, 8 heads/block.
// Writes the transpose(0,2,1,3,4).reshape scramble directly, as bf16 for Wo.
// ---------------------------------------------------------------------------
__global__ void attn_kernel(const float* __restrict__ Q,
                            const float* __restrict__ K,
                            const float* __restrict__ V,
                            __bf16* __restrict__ Abf) {
  __shared__ float sQ[H_][N_ * DK_];
  __shared__ float sK[H_][N_ * DK_];
  __shared__ float sV[H_][N_ * DK_];
  __shared__ float sS[H_][N_ * N_];

  int h = threadIdx.x >> 5, lane = threadIdx.x & 31;
  int bl = blockIdx.x;
  int b = bl / L_, l = bl % L_;
  size_t tok0 = (size_t)bl * N_;

  for (int idx = lane; idx < N_ * DK_; idx += 32) {
    int n = idx / DK_, d = idx % DK_;
    size_t off = (tok0 + n) * C_ + h * DK_ + d;
    sQ[h][idx] = Q[off];
    sK[h][idx] = K[off];
    sV[h][idx] = V[off];
  }
  __syncthreads();

  for (int idx = lane; idx < N_ * N_; idx += 32) {
    int i = idx / N_, j = idx % N_;
    float s = 0.0f;
#pragma unroll
    for (int d = 0; d < DK_; d++) s += sQ[h][i * DK_ + d] * sK[h][j * DK_ + d];
    sS[h][idx] = s * 0.25f;   // 1/sqrt(DK=16)
  }
  __syncthreads();

  if (lane < N_) {
    int i = lane;
    float mx = -1e30f;
    for (int j = 0; j < N_; j++) mx = fmaxf(mx, sS[h][i * N_ + j]);
    float sum = 0.0f;
    for (int j = 0; j < N_; j++) {
      float e = __expf(sS[h][i * N_ + j] - mx);
      sS[h][i * N_ + j] = e;
      sum += e;
    }
    float inv = 1.0f / sum;
    for (int j = 0; j < N_; j++) sS[h][i * N_ + j] *= inv;
  }
  __syncthreads();

  for (int idx = lane; idx < N_ * DK_; idx += 32) {
    int i = idx / DK_, d = idx % DK_;
    float s = 0.0f;
    for (int j = 0; j < N_; j++) s += sS[h][i * N_ + j] * sV[h][j * DK_ + d];
    // scramble: flat (n*L + l) over [N,L] re-viewed as (l2*N + n2) over [L,N]
    int gidx = i * L_ + l;
    int l2 = gidx / N_, n2 = gidx % N_;
    size_t t2 = ((size_t)b * L_ + l2) * N_ + n2;
    Abf[t2 * C_ + h * DK_ + d] = (__bf16)s;
  }
}

// ---------------------------------------------------------------------------
extern "C" void kernel_launch(void* const* d_in, const int* in_sizes, int n_in,
                              void* d_out, int out_size, void* d_ws, size_t ws_size,
                              hipStream_t stream) {
  (void)in_sizes; (void)n_in; (void)out_size; (void)ws_size;
  const float* x    = (const float*)d_in[0];
  const float* Wq   = (const float*)d_in[1];
  const float* bq   = (const float*)d_in[2];
  const float* Wk   = (const float*)d_in[3];
  const float* bk   = (const float*)d_in[4];
  const float* Wv   = (const float*)d_in[5];
  const float* bv   = (const float*)d_in[6];
  const float* Wo   = (const float*)d_in[7];
  const float* bo   = (const float*)d_in[8];
  const float* W1   = (const float*)d_in[9];
  const float* bf1  = (const float*)d_in[10];
  const float* W2   = (const float*)d_in[11];
  const float* bf2  = (const float*)d_in[12];
  const float* ln1g = (const float*)d_in[13];
  const float* ln1b = (const float*)d_in[14];
  const float* lnAg = (const float*)d_in[15];
  const float* lnAb = (const float*)d_in[16];
  const float* lnFg = (const float*)d_in[17];
  const float* lnFb = (const float*)d_in[18];
  const float* peg  = (const float*)d_in[19];
  const float* peb  = (const float*)d_in[20];
  float* h = (float*)d_out;   // hidden state lives in d_out between layers

  // ---- workspace carve-up (all offsets 256B aligned) ----
  char* ws = (char*)d_ws;
  size_t off = 0;
  auto alloc = [&](size_t bytes) -> void* {
    void* p = ws + off;
    off += (bytes + 255) & ~(size_t)255;
    return p;
  };
  const size_t TC4 = (size_t)T_ * C_ * 4;
  const size_t TC2 = (size_t)T_ * C_ * 2;
  float*  F0  = (float*) alloc(TC4);              // y = LN1(h)  (residual base)
  float*  F1  = (float*) alloc(TC4);              // Q
  float*  F2  = (float*) alloc(TC4);              // K
  float*  F3  = (float*) alloc(TC4);              // V
  float*  F4  = (float*) alloc(TC4);              // attn-proj out / ff2 out
  float*  F5  = (float*) alloc(TC4);              // z / u
  __bf16* BF0 = (__bf16*)alloc(TC2);              // bf16(z/u) for FF1
  __bf16* BF1 = (__bf16*)alloc(TC2);              // bf16(y)   for QKV
  __bf16* BF2 = (__bf16*)alloc(TC2);              // bf16(attn out, scrambled)
  __bf16* MBF = (__bf16*)alloc((size_t)T_ * DFF_ * 2);  // mish(FF1) bf16
  __bf16* WqB = (__bf16*)alloc((size_t)NL_ * C_ * C_ * 2);
  __bf16* WkB = (__bf16*)alloc((size_t)NL_ * C_ * C_ * 2);
  __bf16* WvB = (__bf16*)alloc((size_t)NL_ * C_ * C_ * 2);
  __bf16* WoB = (__bf16*)alloc((size_t)NL_ * C_ * C_ * 2);
  __bf16* W1B = (__bf16*)alloc((size_t)NL_ * DFF_ * C_ * 2);
  __bf16* W2B = (__bf16*)alloc((size_t)NL_ * C_ * DFF_ * 2);

  // ---- weights -> bf16 (deterministic, every call) ----
  {
    int nw = NL_ * C_ * C_;
    int nf = NL_ * DFF_ * C_;
    dim3 blk(256);
    f2bf_kernel<<<dim3((nw + 255) / 256), blk, 0, stream>>>(Wq, WqB, nw);
    f2bf_kernel<<<dim3((nw + 255) / 256), blk, 0, stream>>>(Wk, WkB, nw);
    f2bf_kernel<<<dim3((nw + 255) / 256), blk, 0, stream>>>(Wv, WvB, nw);
    f2bf_kernel<<<dim3((nw + 255) / 256), blk, 0, stream>>>(Wo, WoB, nw);
    f2bf_kernel<<<dim3((nf + 255) / 256), blk, 0, stream>>>(W1, W1B, nf);
    f2bf_kernel<<<dim3((nf + 255) / 256), blk, 0, stream>>>(W2, W2B, nf);
  }

  const dim3 lnGrid(T_ / 8), lnBlk(256);
  const dim3 gBlk(256);
  const dim3 gGridC(C_ / 32,  T_ / 256);   // N=128 GEMMs  (4, 176)
  const dim3 gGridF(DFF_ / 32, T_ / 256);  // N=512 GEMM   (16, 176)

  // ---- positional encoding: h = x + LN(x + pe) ----
  pe_ln_kernel<<<lnGrid, lnBlk, 0, stream>>>(x, peg, peb, h);

  for (int l = 0; l < NL_; l++) {
    const __bf16* wq = WqB + (size_t)l * C_ * C_;
    const __bf16* wk = WkB + (size_t)l * C_ * C_;
    const __bf16* wv = WvB + (size_t)l * C_ * C_;
    const __bf16* wo = WoB + (size_t)l * C_ * C_;
    const __bf16* w1 = W1B + (size_t)l * DFF_ * C_;
    const __bf16* w2 = W2B + (size_t)l * C_ * DFF_;

    // y = LN1(h)
    ln_kernel<<<lnGrid, lnBlk, 0, stream>>>(h, nullptr,
        ln1g + l * C_, ln1b + l * C_, F0, BF1);

    // Q, K, V = y @ W^T + b   (WMMA bf16)
    gemm_bf16_kernel<0><<<gGridC, gBlk, 0, stream>>>(BF1, wq, bq + l * C_, F1, nullptr, T_, C_, C_);
    gemm_bf16_kernel<0><<<gGridC, gBlk, 0, stream>>>(BF1, wk, bk + l * C_, F2, nullptr, T_, C_, C_);
    gemm_bf16_kernel<0><<<gGridC, gBlk, 0, stream>>>(BF1, wv, bv + l * C_, F3, nullptr, T_, C_, C_);

    // attention over nodes + scramble -> bf16
    attn_kernel<<<dim3(B_ * L_), dim3(256), 0, stream>>>(F1, F2, F3, BF2);

    // o = a @ Wo^T + bo
    gemm_bf16_kernel<0><<<gGridC, gBlk, 0, stream>>>(BF2, wo, bo + l * C_, F4, nullptr, T_, C_, C_);

    // z = LN(y + o)
    ln_kernel<<<lnGrid, lnBlk, 0, stream>>>(F4, F0,
        lnAg + l * C_, lnAb + l * C_, F5, BF0);

    // two identical feed-forward residual blocks
    for (int p = 0; p < 2; p++) {
      // mish(u @ W1^T + b1) -> bf16
      gemm_bf16_kernel<1><<<gGridF, gBlk, 0, stream>>>(BF0, w1, bf1 + l * DFF_, nullptr, MBF, T_, DFF_, C_);
      // f = m @ W2^T + b2
      gemm_bf16_kernel<0><<<gGridC, gBlk, 0, stream>>>(MBF, w2, bf2 + l * C_, F4, nullptr, T_, C_, DFF_);
      // u' = LN(u + f); last pass of the layer writes the new hidden into d_out
      if (p == 0) {
        ln_kernel<<<lnGrid, lnBlk, 0, stream>>>(F4, F5,
            lnFg + l * C_, lnFb + l * C_, F5, BF0);
      } else {
        ln_kernel<<<lnGrid, lnBlk, 0, stream>>>(F4, F5,
            lnFg + l * C_, lnFb + l * C_, h, nullptr);
      }
    }
  }
}
